// KarpathyTransformerModel_24249385353821
// MI455X (gfx1250) — compile-verified, matrix operand order checked by reading, not run on hardware
//
#include <hip/hip_runtime.h>
#include <math.h>

// ---------------- model dims ----------------
constexpr int Bv  = 2;
constexpr int Tt  = 256;
constexpr int Cc  = 256;
constexpr int C2  = 512;     // 2*C
constexpr int Hh  = 4;
constexpr int HSs = 64;
constexpr int HIDd= 1024;
constexpr int Ll  = 2;
constexpr int Vv  = 8000;

#define USE_ASYNC 1          // flip to 0 if the assembler rejects async-to-LDS mnemonics

typedef __bf16 bf16_t;
typedef __attribute__((ext_vector_type(16))) __bf16 v16bf;
typedef __attribute__((ext_vector_type(8)))  float  v8f;

// ---------------- WMMA helpers ----------------
// CDNA5 16-bit A-frag layout (16x32, MxK): lane holds row m = lane%16,
// elements e<8 at K = (lane/16)*8 + e ; elements e>=8 at K = 16 + (lane/16)*8 + (e-8).
// B-frag (K x 16 cols) mirrors A with n = lane%16 when the tile is stored [n][k].
__device__ __forceinline__ v16bf load_frag(const bf16_t* base, int stride, int lane) {
  const bf16_t* p = base + (lane & 15) * stride + ((lane >> 4) << 3);
  v16bf f;
#pragma unroll
  for (int i = 0; i < 8; ++i) { f[i] = p[i]; f[8 + i] = p[16 + i]; }
  return f;
}

__device__ __forceinline__ v8f wmma_bf16(v16bf a, v16bf b, v8f acc) {
  return __builtin_amdgcn_wmma_f32_16x16x32_bf16(false, a, false, b, (short)0, acc,
                                                 false, false);
}

__device__ __forceinline__ int acc_row(int lane, int i) { return ((lane >> 4) << 3) + i; }

__device__ __forceinline__ float gelu_f(float x) {
  return 0.5f * x * (1.0f + erff(x * 0.7071067811865475f));
}

// ---------------- async global->LDS staging ----------------
// Copies 16B per active lane directly into LDS (ASYNCcnt-tracked, no VGPR bounce).
__device__ __forceinline__ void copy16(bf16_t* lds_dst, const bf16_t* gsrc) {
#if USE_ASYNC
  unsigned lds_off = (unsigned)(unsigned long long)(void*)lds_dst;  // low 32b = LDS offset
  asm volatile("global_load_async_to_lds_b128 %0, %1, off"
               :: "v"(lds_off), "v"((unsigned long long)gsrc)
               : "memory");
#else
  *(uint4*)lds_dst = *(const uint4*)gsrc;
#endif
}
__device__ __forceinline__ void async_wait() {
#if USE_ASYNC
  asm volatile("s_wait_asynccnt 0x0" ::: "memory");
#endif
}

// stage `rows` rows of 32 bf16 (64B) each: dst[r][0..31] = src[r*row_stride + 0..31]
__device__ __forceinline__ void stage_rows(bf16_t* dst, const bf16_t* src,
                                           size_t row_stride, int rows,
                                           int tid, int nthr) {
  for (int e = tid; e < rows * 4; e += nthr) {
    int r = e >> 2, seg = (e & 3) * 8;
    copy16(dst + r * 32 + seg, src + (size_t)r * row_stride + seg);
  }
}

// ---------------- prep: transpose+convert weights to bf16 [N][K] ----------------
__global__ void k_transpose_bf16(const float* __restrict__ src, bf16_t* __restrict__ dst,
                                 int K, int N) {
  size_t base = (size_t)blockIdx.y * K * N;
  int i = blockIdx.x * blockDim.x + threadIdx.x;
  if (i < K * N) {
    int n = i / K, k = i % K;
    dst[base + i] = (bf16_t)src[base + (size_t)k * N + n];
  }
}

// val_w[l]: [H][C][HS] -> vwt[l]: [N=H*HS][K=C]
__global__ void k_prep_valw(const float* __restrict__ vw, bf16_t* __restrict__ dst) {
  size_t base = (size_t)blockIdx.y * Cc * Cc;
  int i = blockIdx.x * blockDim.x + threadIdx.x;   // n*Cc + k
  int n = i >> 8, k = i & 255;
  dst[base + i] = (bf16_t)vw[base + (size_t)(n >> 6) * Cc * HSs + (size_t)k * HSs + (n & 63)];
}

// ---------------- kernels ----------------

__global__ void k_embed(const int* __restrict__ idx, const float* __restrict__ tok_emb,
                        float* __restrict__ x, bf16_t* __restrict__ xb) {
  int bt = blockIdx.x, c = threadIdx.x;
  float v = tok_emb[(size_t)idx[bt] * Cc + c];
  x[(size_t)bt * Cc + c]  = v;
  xb[(size_t)bt * Cc + c] = (bf16_t)v;
}

// hmid[r,n] = gelu( pos_table[dist(r)] @ pos_w1 ) ; M=T*T, K=C, N=2C
__global__ void __launch_bounds__(256) k_pe1(const float* __restrict__ pos_table,
                                             const bf16_t* __restrict__ w1t,  // [2C][C]
                                             bf16_t* __restrict__ hmid) {
  __shared__ alignas(16) bf16_t As[16 * 32];
  __shared__ alignas(16) bf16_t Bs[C2 * 32];
  int row0 = blockIdx.x * 16;
  int tid = threadIdx.x, lane = tid & 31, wave = tid >> 5;
  v8f acc[4] = {};
  for (int k0 = 0; k0 < Cc; k0 += 32) {
    for (int e = tid; e < 16 * 32; e += 256) {       // gathered f32 -> bf16 (VALU)
      int m = e >> 5, kk = e & 31;
      int r = row0 + m, t = r >> 8, s = r & 255;
      int d = (s <= t) ? (t - s) : 0;
      As[e] = (bf16_t)pos_table[(size_t)d * Cc + k0 + kk];
    }
    stage_rows(Bs, w1t + k0, Cc, C2, tid, 256);      // async DMA
    async_wait();
    __syncthreads();
    v16bf a = load_frag(As, 32, lane);
#pragma unroll
    for (int j = 0; j < 4; ++j) {
      v16bf bfrag = load_frag(Bs + (wave * 64 + j * 16) * 32, 32, lane);
      acc[j] = wmma_bf16(a, bfrag, acc[j]);
    }
    __syncthreads();
  }
#pragma unroll
  for (int j = 0; j < 4; ++j)
#pragma unroll
    for (int i = 0; i < 8; ++i) {
      int m = acc_row(lane, i), n = wave * 64 + j * 16 + (lane & 15);
      hmid[(size_t)(row0 + m) * C2 + n] = (bf16_t)gelu_f(acc[j][i]);
    }
}

// pe[r,:] = LN( hmid[r,:] @ pos_w2 + pos_b2 ) ; M=T*T, K=2C, N=2C, row-LN fused
union PeU { bf16_t B[C2 * 32]; float Ct[16 * C2]; };
__global__ void __launch_bounds__(256) k_pe2(const bf16_t* __restrict__ hmid,
                                             const bf16_t* __restrict__ w2t,  // [2C][2C]
                                             const float* __restrict__ b2,
                                             const float* __restrict__ ln_g,
                                             const float* __restrict__ ln_b,
                                             bf16_t* __restrict__ pe) {
  __shared__ alignas(16) bf16_t As[16 * 32];
  __shared__ alignas(16) PeU u;
  __shared__ float mu_s[16], rs_s[16];
  int row0 = blockIdx.x * 16;
  int tid = threadIdx.x, lane = tid & 31, wave = tid >> 5;
  v8f acc[4] = {};
  for (int k0 = 0; k0 < C2; k0 += 32) {
    stage_rows(As, hmid + (size_t)row0 * C2 + k0, C2, 16, tid, 256);
    stage_rows(u.B, w2t + k0, C2, C2, tid, 256);
    async_wait();
    __syncthreads();
    v16bf a = load_frag(As, 32, lane);
#pragma unroll
    for (int j = 0; j < 4; ++j) {
      v16bf bfrag = load_frag(u.B + (wave * 64 + j * 16) * 32, 32, lane);
      acc[j] = wmma_bf16(a, bfrag, acc[j]);
    }
    __syncthreads();
  }
#pragma unroll
  for (int j = 0; j < 4; ++j)
#pragma unroll
    for (int i = 0; i < 8; ++i) {
      int m = acc_row(lane, i), n = wave * 64 + j * 16 + (lane & 15);
      u.Ct[m * C2 + n] = acc[j][i] + b2[n];
    }
  __syncthreads();
  if (tid < 16) {
    float s = 0.f, s2 = 0.f;
    for (int n = 0; n < C2; ++n) { float v = u.Ct[tid * C2 + n]; s += v; s2 += v * v; }
    float mu = s / C2;
    mu_s[tid] = mu;
    rs_s[tid] = rsqrtf(s2 / C2 - mu * mu + 1e-5f);
  }
  __syncthreads();
  for (int e = tid; e < 16 * C2; e += 256) {
    int m = e >> 9, n = e & 511;
    float v = (u.Ct[e] - mu_s[m]) * rs_s[m] * ln_g[n] + ln_b[n];
    pe[(size_t)(row0 + m) * C2 + n] = (bf16_t)v;
  }
}

// wei[b,h,t,s] = ( att_w2[h] . gelu( (concat(x[b,s],x[b,t]) + pe[t,s]) @ att_w1[h] ) + b2[h] ) / 16
__global__ void __launch_bounds__(256) k_score(const bf16_t* __restrict__ xb,
                                               const bf16_t* __restrict__ pe,
                                               const bf16_t* __restrict__ w1t_l, // [H][C][2C]
                                               const float* __restrict__ w2_l,
                                               const float* __restrict__ b2_l,
                                               float* __restrict__ wei) {
  int h = blockIdx.y, b = blockIdx.z;
  int row0 = blockIdx.x * 16;     // index into T*T; 16 rows share t
  int t = row0 >> 8, s0 = row0 & 255;
  const bf16_t* W1t = w1t_l + (size_t)h * Cc * C2;
  const float*  W2  = w2_l + (size_t)h * Cc;
  __shared__ alignas(16) bf16_t As[16 * 32];
  __shared__ alignas(16) bf16_t Bs[Cc * 32];
  __shared__ float wsum[16];
  int tid = threadIdx.x, lane = tid & 31, wave = tid >> 5;
  if (tid < 16) wsum[tid] = 0.f;
  v8f acc[2] = {};
  const bf16_t* xrow_t = xb + ((size_t)b * Tt + t) * Cc;
  for (int k0 = 0; k0 < C2; k0 += 32) {
    for (int e = tid; e < 16 * 32; e += 256) {       // computed A tile (VALU)
      int m = e >> 5, kk = e & 31, k = k0 + kk, s = s0 + m;
      float xa = (k < Cc) ? (float)xb[((size_t)b * Tt + s) * Cc + k]
                          : (float)xrow_t[k - Cc];
      float pv = (float)pe[((size_t)t * Tt + s) * C2 + k];
      As[e] = (bf16_t)(xa + pv);
    }
    stage_rows(Bs, W1t + k0, C2, Cc, tid, 256);      // async DMA
    async_wait();
    __syncthreads();
    v16bf a = load_frag(As, 32, lane);
#pragma unroll
    for (int j = 0; j < 2; ++j) {
      v16bf bfrag = load_frag(Bs + (wave * 32 + j * 16) * 32, 32, lane);
      acc[j] = wmma_bf16(a, bfrag, acc[j]);
    }
    __syncthreads();
  }
#pragma unroll
  for (int j = 0; j < 2; ++j)
#pragma unroll
    for (int i = 0; i < 8; ++i) {
      int m = acc_row(lane, i), n = wave * 32 + j * 16 + (lane & 15);
      atomicAdd(&wsum[m], gelu_f(acc[j][i]) * W2[n]);
    }
  __syncthreads();
  if (tid < 16) {
    float v = (wsum[tid] + b2_l[h]) * 0.0625f;   // C^-0.5 = 1/16
    wei[(((size_t)b * Hh + h) * Tt + t) * Tt + s0 + tid] = v;
  }
}

// causal softmax over s; masked entries -> 0 in bf16 prob matrix
__global__ void k_softmax(const float* __restrict__ wei, bf16_t* __restrict__ p) {
  int t = blockIdx.x, h = blockIdx.y, b = blockIdx.z, s = threadIdx.x;
  size_t base = (((size_t)b * Hh + h) * Tt + t) * Tt;
  float v = (s <= t) ? wei[base + s] : -3.0e38f;
  __shared__ float red[256];
  red[s] = v; __syncthreads();
  for (int off = 128; off > 0; off >>= 1) { if (s < off) red[s] = fmaxf(red[s], red[s + off]); __syncthreads(); }
  float mx = red[0]; __syncthreads();
  float e = (s <= t) ? __expf(v - mx) : 0.f;
  red[s] = e; __syncthreads();
  for (int off = 128; off > 0; off >>= 1) { if (s < off) red[s] += red[s + off]; __syncthreads(); }
  p[base + s] = (bf16_t)(e / red[0]);
}

// vbufT[b][h*HS+d][s] = x[b,s,:] @ val_w[l,h,:,d] ; M=B*T, K=C, N=H*HS
__global__ void __launch_bounds__(256) k_vgemm(const bf16_t* __restrict__ xb,
                                               const bf16_t* __restrict__ vwt_l, // [C][C]
                                               bf16_t* __restrict__ vbufT) {
  __shared__ alignas(16) bf16_t As[16 * 32];
  __shared__ alignas(16) bf16_t Bs[Cc * 32];
  int row0 = blockIdx.x * 16;
  int tid = threadIdx.x, lane = tid & 31, wave = tid >> 5;
  v8f acc[2] = {};
  for (int k0 = 0; k0 < Cc; k0 += 32) {
    stage_rows(As, xb + (size_t)row0 * Cc + k0, Cc, 16, tid, 256);
    stage_rows(Bs, vwt_l + k0, Cc, Cc, tid, 256);
    async_wait();
    __syncthreads();
    v16bf a = load_frag(As, 32, lane);
#pragma unroll
    for (int j = 0; j < 2; ++j) {
      v16bf bfrag = load_frag(Bs + (wave * 32 + j * 16) * 32, 32, lane);
      acc[j] = wmma_bf16(a, bfrag, acc[j]);
    }
    __syncthreads();
  }
#pragma unroll
  for (int j = 0; j < 2; ++j)
#pragma unroll
    for (int i = 0; i < 8; ++i) {
      int m = acc_row(lane, i), n = wave * 32 + j * 16 + (lane & 15);
      int r = row0 + m;                      // r = b*T + s
      vbufT[((size_t)(r >> 8) * Cc + n) * Tt + (r & 255)] = (bf16_t)acc[j][i];
    }
}

// attout[b,t, h*HS+d] = sum_s p[b,h,t,s] * v[b,s,h*HS+d] ; per (b,h): M=T,K=T,N=HS
__global__ void __launch_bounds__(128) k_av(const bf16_t* __restrict__ p,
                                            const bf16_t* __restrict__ vbufT,
                                            bf16_t* __restrict__ attout) {
  int h = blockIdx.y, b = blockIdx.z, t0 = blockIdx.x * 16;
  __shared__ alignas(16) bf16_t As[16 * 32];
  __shared__ alignas(16) bf16_t Bs[64 * 32];
  int tid = threadIdx.x, lane = tid & 31, wave = tid >> 5;
  v8f acc = {};
  size_t pbase = (((size_t)b * Hh + h) * Tt) * (size_t)Tt;
  const bf16_t* vrows = vbufT + ((size_t)b * Cc + h * HSs) * Tt;
  for (int k0 = 0; k0 < Tt; k0 += 32) {
    stage_rows(As, p + pbase + (size_t)t0 * Tt + k0, Tt, 16, tid, 128);
    stage_rows(Bs, vrows + k0, Tt, 64, tid, 128);
    async_wait();
    __syncthreads();
    v16bf a = load_frag(As, 32, lane);
    v16bf bfrag = load_frag(Bs + (wave * 16) * 32, 32, lane);
    acc = wmma_bf16(a, bfrag, acc);
    __syncthreads();
  }
#pragma unroll
  for (int i = 0; i < 8; ++i) {
    int m = acc_row(lane, i), n = wave * 16 + (lane & 15);
    attout[((size_t)b * Tt + t0 + m) * Cc + h * HSs + n] = (bf16_t)acc[i];
  }
}

// x += attout @ proj_w + proj_b
__global__ void __launch_bounds__(256) k_proj(const bf16_t* __restrict__ attout,
                                              const bf16_t* __restrict__ pwt, // [C][C]
                                              const float* __restrict__ pb,
                                              float* __restrict__ x, bf16_t* __restrict__ xb) {
  __shared__ alignas(16) bf16_t As[16 * 32];
  __shared__ alignas(16) bf16_t Bs[Cc * 32];
  int row0 = blockIdx.x * 16;
  int tid = threadIdx.x, lane = tid & 31, wave = tid >> 5;
  v8f acc[2] = {};
  for (int k0 = 0; k0 < Cc; k0 += 32) {
    stage_rows(As, attout + (size_t)row0 * Cc + k0, Cc, 16, tid, 256);
    stage_rows(Bs, pwt + k0, Cc, Cc, tid, 256);
    async_wait();
    __syncthreads();
    v16bf a = load_frag(As, 32, lane);
#pragma unroll
    for (int j = 0; j < 2; ++j) {
      v16bf bfrag = load_frag(Bs + (wave * 32 + j * 16) * 32, 32, lane);
      acc[j] = wmma_bf16(a, bfrag, acc[j]);
    }
    __syncthreads();
  }
#pragma unroll
  for (int j = 0; j < 2; ++j)
#pragma unroll
    for (int i = 0; i < 8; ++i) {
      int m = acc_row(lane, i), n = wave * 32 + j * 16 + (lane & 15);
      size_t ix = (size_t)(row0 + m) * Cc + n;
      float v = x[ix] + acc[j][i] + pb[n];
      x[ix] = v;  xb[ix] = (bf16_t)v;
    }
}

// row LayerNorm over C=256, output bf16
__global__ void k_ln(const float* __restrict__ x, const float* __restrict__ g,
                     const float* __restrict__ bta, bf16_t* __restrict__ out) {
  int r = blockIdx.x, c = threadIdx.x;
  float v = x[(size_t)r * Cc + c];
  __shared__ float red[256];
  red[c] = v; __syncthreads();
  for (int off = 128; off > 0; off >>= 1) { if (c < off) red[c] += red[c + off]; __syncthreads(); }
  float mu = red[0] / Cc; __syncthreads();
  float d = v - mu;
  red[c] = d * d; __syncthreads();
  for (int off = 128; off > 0; off >>= 1) { if (c < off) red[c] += red[c + off]; __syncthreads(); }
  out[(size_t)r * Cc + c] = (bf16_t)(d * rsqrtf(red[0] / Cc + 1e-5f) * g[c] + bta[c]);
}

// f1 = gelu(hb @ ff_w1) ; grid.y splits N=HID into 512-wide chunks
__global__ void __launch_bounds__(256) k_ff1(const bf16_t* __restrict__ hb,
                                             const bf16_t* __restrict__ w1t, // [HID][C]
                                             bf16_t* __restrict__ f1) {
  __shared__ alignas(16) bf16_t As[16 * 32];
  __shared__ alignas(16) bf16_t Bs[512 * 32];
  int row0 = blockIdx.x * 16, ncol0 = blockIdx.y * 512;
  int tid = threadIdx.x, lane = tid & 31, wave = tid >> 5;
  v8f acc[4] = {};
  for (int k0 = 0; k0 < Cc; k0 += 32) {
    stage_rows(As, hb + (size_t)row0 * Cc + k0, Cc, 16, tid, 256);
    stage_rows(Bs, w1t + (size_t)ncol0 * Cc + k0, Cc, 512, tid, 256);
    async_wait();
    __syncthreads();
    v16bf a = load_frag(As, 32, lane);
#pragma unroll
    for (int j = 0; j < 4; ++j) {
      v16bf bfrag = load_frag(Bs + (wave * 64 + j * 16) * 32, 32, lane);
      acc[j] = wmma_bf16(a, bfrag, acc[j]);
    }
    __syncthreads();
  }
#pragma unroll
  for (int j = 0; j < 4; ++j)
#pragma unroll
    for (int i = 0; i < 8; ++i) {
      int m = acc_row(lane, i), n = ncol0 + wave * 64 + j * 16 + (lane & 15);
      f1[(size_t)(row0 + m) * HIDd + n] = (bf16_t)gelu_f(acc[j][i]);
    }
}

// x += f1 @ ff_w2 + ff_b2 ; K=HID
__global__ void __launch_bounds__(256) k_ff2(const bf16_t* __restrict__ f1,
                                             const bf16_t* __restrict__ w2t, // [C][HID]
                                             const float* __restrict__ b2,
                                             float* __restrict__ x, bf16_t* __restrict__ xb) {
  __shared__ alignas(16) bf16_t As[16 * 32];
  __shared__ alignas(16) bf16_t Bs[Cc * 32];
  int row0 = blockIdx.x * 16;
  int tid = threadIdx.x, lane = tid & 31, wave = tid >> 5;
  v8f acc[2] = {};
  for (int k0 = 0; k0 < HIDd; k0 += 32) {
    stage_rows(As, f1 + (size_t)row0 * HIDd + k0, HIDd, 16, tid, 256);
    stage_rows(Bs, w2t + k0, HIDd, Cc, tid, 256);
    async_wait();
    __syncthreads();
    v16bf a = load_frag(As, 32, lane);
#pragma unroll
    for (int j = 0; j < 2; ++j) {
      v16bf bfrag = load_frag(Bs + (wave * 32 + j * 16) * 32, 32, lane);
      acc[j] = wmma_bf16(a, bfrag, acc[j]);
    }
    __syncthreads();
  }
#pragma unroll
  for (int j = 0; j < 2; ++j)
#pragma unroll
    for (int i = 0; i < 8; ++i) {
      int m = acc_row(lane, i), n = wave * 32 + j * 16 + (lane & 15);
      size_t ix = (size_t)(row0 + m) * Cc + n;
      float v = x[ix] + acc[j][i] + b2[n];
      x[ix] = v;  xb[ix] = (bf16_t)v;
    }
}

// logits = xf @ head_w + head_b ; grid.y splits N=V into 256-wide chunks (tail guarded)
__global__ void __launch_bounds__(256) k_head(const bf16_t* __restrict__ xf,
                                              const bf16_t* __restrict__ hwt, // [V][C]
                                              const float* __restrict__ hbias,
                                              float* __restrict__ out) {
  __shared__ alignas(16) bf16_t As[16 * 32];
  __shared__ alignas(16) bf16_t Bs[256 * 32];
  int row0 = blockIdx.x * 16, ncol0 = blockIdx.y * 256;
  int tid = threadIdx.x, lane = tid & 31, wave = tid >> 5;
  v8f acc[2] = {};
  for (int k0 = 0; k0 < Cc; k0 += 32) {
    stage_rows(As, xf + (size_t)row0 * Cc + k0, Cc, 16, tid, 256);
    for (int e = tid; e < 256 * 4; e += 256) {
      int n = e >> 2, seg = (e & 3) * 8, gn = ncol0 + n;
      if (gn < Vv) copy16(Bs + n * 32 + seg, hwt + (size_t)gn * Cc + k0 + seg);
      else { uint4 z = {0, 0, 0, 0}; *(uint4*)(Bs + n * 32 + seg) = z; }
    }
    async_wait();
    __syncthreads();
    v16bf a = load_frag(As, 32, lane);
#pragma unroll
    for (int j = 0; j < 2; ++j) {
      v16bf bfrag = load_frag(Bs + (wave * 32 + j * 16) * 32, 32, lane);
      acc[j] = wmma_bf16(a, bfrag, acc[j]);
    }
    __syncthreads();
  }
#pragma unroll
  for (int j = 0; j < 2; ++j)
#pragma unroll
    for (int i = 0; i < 8; ++i) {
      int m = acc_row(lane, i), n = ncol0 + wave * 32 + j * 16 + (lane & 15);
      if (n < Vv) out[(size_t)(row0 + m) * Vv + n] = acc[j][i] + hbias[n];
    }
}

// ---------------- launch ----------------
extern "C" void kernel_launch(void* const* d_in, const int* in_sizes, int n_in,
                              void* d_out, int out_size, void* d_ws, size_t ws_size,
                              hipStream_t stream) {
  (void)in_sizes; (void)n_in; (void)out_size; (void)ws_size;
  const int*   idx      = (const int*)  d_in[0];
  const float* tok_emb  = (const float*)d_in[1];
  const float* pos_table= (const float*)d_in[2];
  const float* pos_w1   = (const float*)d_in[3];
  const float* pos_w2   = (const float*)d_in[4];
  const float* pos_b2   = (const float*)d_in[5];
  const float* pos_ln_g = (const float*)d_in[6];
  const float* pos_ln_b = (const float*)d_in[7];
  const float* att_w1   = (const float*)d_in[8];
  const float* att_w2   = (const float*)d_in[9];
  const float* att_b2   = (const float*)d_in[10];
  const float* val_w    = (const float*)d_in[11];
  const float* proj_w   = (const float*)d_in[12];
  const float* proj_b   = (const float*)d_in[13];
  const float* ln2_g    = (const float*)d_in[14];
  const float* ln2_b    = (const float*)d_in[15];
  const float* ff_w1    = (const float*)d_in[16];
  const float* ff_w2    = (const float*)d_in[17];
  const float* ff_b2    = (const float*)d_in[18];
  const float* lnf_g    = (const float*)d_in[19];
  const float* lnf_b    = (const float*)d_in[20];
  const float* head_w   = (const float*)d_in[21];
  const float* head_b   = (const float*)d_in[22];
  float* logits = (float*)d_out;

  char* ws = (char*)d_ws;
  size_t off = 0;
  auto take = [&](size_t bytes) -> char* {
    char* pp = ws + off;
    off += (bytes + 255) & ~(size_t)255;
    return pp;
  };
  float*  x      = (float*) take((size_t)Bv * Tt * Cc * 4);
  bf16_t* xb     = (bf16_t*)take((size_t)Bv * Tt * Cc * 2);
  bf16_t* hmid   = (bf16_t*)take((size_t)Tt * Tt * C2 * 2);
  bf16_t* pe     = (bf16_t*)take((size_t)Tt * Tt * C2 * 2);
  float*  wei    = (float*) take((size_t)Bv * Hh * Tt * Tt * 4);
  bf16_t* pmat   = (bf16_t*)take((size_t)Bv * Hh * Tt * Tt * 2);
  bf16_t* vbufT  = (bf16_t*)take((size_t)Bv * Tt * Cc * 2);
  bf16_t* attout = (bf16_t*)take((size_t)Bv * Tt * Cc * 2);
  bf16_t* hb     = (bf16_t*)take((size_t)Bv * Tt * Cc * 2);
  bf16_t* f1     = (bf16_t*)take((size_t)Bv * Tt * HIDd * 2);
  bf16_t* xf     = (bf16_t*)take((size_t)Bv * Tt * Cc * 2);
  // bf16 [N][K] transposed weights
  bf16_t* w1t    = (bf16_t*)take((size_t)C2 * Cc * 2);
  bf16_t* w2t    = (bf16_t*)take((size_t)C2 * C2 * 2);
  bf16_t* aw1t   = (bf16_t*)take((size_t)Ll * Hh * Cc * C2 * 2);
  bf16_t* vwt    = (bf16_t*)take((size_t)Ll * Cc * Cc * 2);
  bf16_t* pwt    = (bf16_t*)take((size_t)Ll * Cc * Cc * 2);
  bf16_t* f1wt   = (bf16_t*)take((size_t)Ll * HIDd * Cc * 2);
  bf16_t* f2wt   = (bf16_t*)take((size_t)Ll * Cc * HIDd * 2);
  bf16_t* hwt    = (bf16_t*)take((size_t)Vv * Cc * 2);

  // ---- prep: weights -> bf16 [N][K] ----
  k_transpose_bf16<<<dim3((Cc * C2 + 255) / 256, 1),       256, 0, stream>>>(pos_w1, w1t, Cc, C2);
  k_transpose_bf16<<<dim3((C2 * C2 + 255) / 256, 1),       256, 0, stream>>>(pos_w2, w2t, C2, C2);
  k_transpose_bf16<<<dim3((C2 * Cc + 255) / 256, Ll * Hh), 256, 0, stream>>>(att_w1, aw1t, C2, Cc);
  k_prep_valw     <<<dim3(Cc * Cc / 256, Ll),              256, 0, stream>>>(val_w, vwt);
  k_transpose_bf16<<<dim3((Cc * Cc + 255) / 256, Ll),      256, 0, stream>>>(proj_w, pwt, Cc, Cc);
  k_transpose_bf16<<<dim3((Cc * HIDd + 255) / 256, Ll),    256, 0, stream>>>(ff_w1, f1wt, Cc, HIDd);
  k_transpose_bf16<<<dim3((HIDd * Cc + 255) / 256, Ll),    256, 0, stream>>>(ff_w2, f2wt, HIDd, Cc);
  k_transpose_bf16<<<dim3((Cc * Vv + 255) / 256, 1),       256, 0, stream>>>(head_w, hwt, Cc, Vv);

  const int MROWS = Bv * Tt;                // 512
  k_embed<<<dim3(MROWS), dim3(Cc), 0, stream>>>(idx, tok_emb, x, xb);
  k_pe1  <<<dim3(Tt * Tt / 16), dim3(256), 0, stream>>>(pos_table, w1t, hmid);
  k_pe2  <<<dim3(Tt * Tt / 16), dim3(256), 0, stream>>>(hmid, w2t, pos_b2,
                                                        pos_ln_g, pos_ln_b, pe);
  for (int l = 0; l < Ll; ++l) {
    k_score  <<<dim3(Tt * Tt / 16, Hh, Bv), dim3(256), 0, stream>>>(
        xb, pe, aw1t + (size_t)l * Hh * Cc * C2, att_w2 + (size_t)l * Hh * Cc,
        att_b2 + (size_t)l * Hh, wei);
    k_softmax<<<dim3(Tt, Hh, Bv), dim3(Tt), 0, stream>>>(wei, pmat);
    k_vgemm  <<<dim3(MROWS / 16), dim3(256), 0, stream>>>(
        xb, vwt + (size_t)l * Cc * Cc, vbufT);
    k_av     <<<dim3(Tt / 16, Hh, Bv), dim3(128), 0, stream>>>(pmat, vbufT, attout);
    k_proj   <<<dim3(MROWS / 16), dim3(256), 0, stream>>>(
        attout, pwt + (size_t)l * Cc * Cc, proj_b + (size_t)l * Cc, x, xb);
    k_ln     <<<dim3(MROWS), dim3(Cc), 0, stream>>>(
        x, ln2_g + (size_t)l * Cc, ln2_b + (size_t)l * Cc, hb);
    k_ff1    <<<dim3(MROWS / 16, HIDd / 512), dim3(256), 0, stream>>>(
        hb, f1wt + (size_t)l * HIDd * Cc, f1);
    k_ff2    <<<dim3(MROWS / 16), dim3(256), 0, stream>>>(
        f1, f2wt + (size_t)l * Cc * HIDd, ff_b2 + (size_t)l * Cc, x, xb);
  }
  k_ln  <<<dim3(MROWS), dim3(Cc), 0, stream>>>(x, lnf_g, lnf_b, xf);
  k_head<<<dim3(MROWS / 16, (Vv + 255) / 256), dim3(256), 0, stream>>>(
      xf, head_w ? hwt : hwt, head_b, logits);
}
